// HQQGroupedGemm_37357625540844
// MI455X (gfx1250) — matrix operand order repaired and checked by reading, not compiled, not run on hardware
//
#include <hip/hip_runtime.h>

typedef __attribute__((ext_vector_type(16))) _Float16 v16h;
typedef __attribute__((ext_vector_type(8)))  _Float16 v8h;
typedef __attribute__((ext_vector_type(8)))  float    v8f;

#define NEXP 8
#define TTOK 2048
#define INF_ 2048
#define OUTF 2048
#define GSZ  64
#define NGRP (INF_ / GSZ)   // 32

#define BM 128
#define BN 128
#define BK 32
#define PAD 8
#define LDA (BK + PAD)      // 40 halves = 80B row stride -> conflict-free strided access
#define LDB (BK + PAD)

template <bool FULL>
__device__ __forceinline__
void gemm_body(const float* __restrict__ x,
               const int*   __restrict__ qw,
               const float* __restrict__ sz,
               float*       __restrict__ out,
               int e, int m0, int n0, int row_start, int rows_e,
               _Float16 (*As)[BM][LDA], _Float16 (*Bs)[BN][LDB])
{
    const int t    = threadIdx.x;
    const int lane = t & 31;
    const int half = lane >> 4;           // wave32 WMMA lane halves
    const int l16  = lane & 15;
    const int w    = t >> 5;              // wave id 0..7
    const int wm   = (w & 3) * 32;        // 4 waves along M  -> 128
    const int wn   = (w >> 2) * 64;       // 2 waves along N  -> 128; 32x64 per wave

    // A staging: thread owns 16 contiguous k of one row
    const int ar = t >> 1;                // row 0..127
    const int ak = (t & 1) * 16;          // k half-row: 0 or 16
    // B staging: thread owns one column n (0..127), 16 consecutive k
    const int bn = t & 127;
    const int bk = (t >> 7) * 16;

    v8f acc[2][4] = {};

    float4   aReg[4];
    int      qReg[16];
    _Float16 hS = (_Float16)0.f, hZ = (_Float16)0.f;

    auto load_tile = [&](int kt) {
        const int k0 = kt * BK;
        // A: 128x32 floats; thread pair covers one 128B row segment -> coalesced
        if (FULL || (m0 + ar < rows_e)) {
            const float* xp = &x[(size_t)(row_start + m0 + ar) * INF_ + k0 + ak];
#pragma unroll
            for (int i = 0; i < 4; ++i) aReg[i] = *(const float4*)(xp + i * 4);
        } else {
#pragma unroll
            for (int i = 0; i < 4; ++i) aReg[i] = make_float4(0.f, 0.f, 0.f, 0.f);
        }
        // B: 32x128 int32 codes; each per-j load is 128B-coalesced across a wave
#pragma unroll
        for (int j = 0; j < 16; ++j) {
            qReg[j] = qw[((size_t)e * INF_ + (size_t)(k0 + bk + j)) * OUTF + n0 + bn];
        }
        // one quant group per BK block: g = k0/GSZ is block-uniform
        const int g = k0 / GSZ;
        const float2 p = *(const float2*)&sz[(((size_t)e * NGRP + g) * OUTF + n0 + bn) * 2];
        hS = (_Float16)p.x; hZ = (_Float16)p.y;
    };

    auto store_tile = [&](int buf) {
        v8h a0, a1;
#pragma unroll
        for (int i = 0; i < 4; ++i) {
            const _Float16 e0 = (_Float16)aReg[i].x, e1 = (_Float16)aReg[i].y;
            const _Float16 e2 = (_Float16)aReg[i].z, e3 = (_Float16)aReg[i].w;
            if (i < 2) { a0[i*4+0]=e0; a0[i*4+1]=e1; a0[i*4+2]=e2; a0[i*4+3]=e3; }
            else { a1[(i-2)*4+0]=e0; a1[(i-2)*4+1]=e1; a1[(i-2)*4+2]=e2; a1[(i-2)*4+3]=e3; }
        }
        *(v8h*)&As[buf][ar][ak]     = a0;   // 16B aligned: ar*80 + {0,32}
        *(v8h*)&As[buf][ar][ak + 8] = a1;
        // dequant in f16: (q-8) exact in f16 -> v_fma_f16
        v8h w0, w1;
#pragma unroll
        for (int j = 0; j < 16; ++j) {
            const _Float16 qh = (_Float16)(qReg[j] - 8);
            const _Float16 v  = qh * hS + hZ;
            if (j < 8) w0[j]     = v;
            else       w1[j - 8] = v;
        }
        *(v8h*)&Bs[buf][bn][bk]     = w0;   // 16B aligned: bn*80 + {0,32}
        *(v8h*)&Bs[buf][bn][bk + 8] = w1;
    };

    auto compute = [&](int buf) {
        v16h af[2], bf[4];
        // A fragment (16-bit A 16x32): lane half h holds K = h*16..+15
#pragma unroll
        for (int mi = 0; mi < 2; ++mi) {
            const int m = wm + mi * 16 + l16;
            const v8h a0 = *(const v8h*)&As[buf][m][half * 16];
            const v8h a1 = *(const v8h*)&As[buf][m][half * 16 + 8];
#pragma unroll
            for (int j = 0; j < 8; ++j) { af[mi][j] = a0[j]; af[mi][j + 8] = a1[j]; }
        }
        // B fragment (32x16): lanes 0-15 hold K=0..15, lanes 16-31 hold K=16..31
#pragma unroll
        for (int ni = 0; ni < 4; ++ni) {
            const int n = wn + ni * 16 + l16;
            const v8h b0 = *(const v8h*)&Bs[buf][n][half * 16];
            const v8h b1 = *(const v8h*)&Bs[buf][n][half * 16 + 8];
#pragma unroll
            for (int j = 0; j < 8; ++j) { bf[ni][j] = b0[j]; bf[ni][j + 8] = b1[j]; }
        }
#pragma unroll
        for (int mi = 0; mi < 2; ++mi)
#pragma unroll
            for (int ni = 0; ni < 4; ++ni)
                acc[mi][ni] = __builtin_amdgcn_wmma_f32_16x16x32_f16(
                    false, af[mi], false, bf[ni], (short)0, acc[mi][ni], false, false);
    };

    const int KT = INF_ / BK;  // 64
    load_tile(0);
    store_tile(0);
    for (int kt = 0; kt < KT; ++kt) {
        const int cur = kt & 1;
        if (kt + 1 < KT) load_tile(kt + 1);   // global loads in flight during WMMAs
        __syncthreads();                      // buf[cur] filled; buf[cur^1] readers done
        compute(cur);
        if (kt + 1 < KT) store_tile(cur ^ 1); // write other buffer, no barrier needed
    }

    // C/D layout: lane half h holds rows M = h*8 + r in VGPR r, N = lane&15
#pragma unroll
    for (int mi = 0; mi < 2; ++mi) {
#pragma unroll
        for (int ni = 0; ni < 4; ++ni) {
            const int col = n0 + wn + ni * 16 + l16;
#pragma unroll
            for (int r = 0; r < 8; ++r) {
                const int lr = wm + mi * 16 + half * 8 + r;
                if (FULL || (m0 + lr < rows_e)) {
                    out[(size_t)(row_start + m0 + lr) * OUTF + col] = acc[mi][ni][r];
                }
            }
        }
    }
}

__global__ __launch_bounds__(256)
void hqq_grouped_gemm_wmma(const float* __restrict__ x,
                           const int*   __restrict__ qw,
                           const float* __restrict__ sz,
                           const int*   __restrict__ tpe,
                           float*       __restrict__ out)
{
    // double-buffered tiles: 2*(128*40 + 128*40)*2B = 40 KB of the 320 KB WGP LDS
    __shared__ _Float16 As[2][BM][LDA];   // x tile, [m][k] f16
    __shared__ _Float16 Bs[2][BN][LDB];   // dequantized W tile, TRANSPOSED [n][k] f16

    const int e  = blockIdx.z;
    const int m0 = blockIdx.y * BM;
    const int n0 = blockIdx.x * BN;

    // expert row range from tokens_per_expert (block-uniform scalar work)
    int row_start = 0;
#pragma unroll
    for (int i = 0; i < NEXP; ++i) {
        int c = tpe[i];
        if (i < e) row_start += c;
    }
    const int rows_e = tpe[e];
    if (m0 >= rows_e) return;             // block-uniform exit: EXEC stays all-1s for WMMA

    if (m0 + BM <= rows_e) {
        gemm_body<true >(x, qw, sz, out, e, m0, n0, row_start, rows_e, As, Bs);
    } else {
        gemm_body<false>(x, qw, sz, out, e, m0, n0, row_start, rows_e, As, Bs);
    }
}

extern "C" void kernel_launch(void* const* d_in, const int* in_sizes, int n_in,
                              void* d_out, int out_size, void* d_ws, size_t ws_size,
                              hipStream_t stream) {
    (void)in_sizes; (void)n_in; (void)out_size; (void)d_ws; (void)ws_size;
    const float* x   = (const float*)d_in[0];
    const int*   qw  = (const int*)d_in[1];
    const float* sz  = (const float*)d_in[2];
    const int*   tpe = (const int*)d_in[3];
    float*       out = (float*)d_out;

    dim3 grid(OUTF / BN, (TTOK + BM - 1) / BM, NEXP);  // 16 x 16 x 8
    hqq_grouped_gemm_wmma<<<grid, 256, 0, stream>>>(x, qw, sz, tpe, out);
}
